// LstmEncoder_19104014533401
// MI455X (gfx1250) — matrix operand order repaired and checked by reading, not compiled, not run on hardware
//
#include <hip/hip_runtime.h>
#include <hip/hip_bf16.h>

// ---------------------------------------------------------------------------
// LSTM encoder for MI455X (gfx1250, wave32, WMMA).
//
//  - x and all weights converted to bf16 once; weights pre-packed into the
//    CDNA5 WMMA B-fragment lane layout (one contiguous 32B/lane load, L2-hot).
//  - Three large WMMA GEMMs compute xw = X@W + b up front (parallel part).
//  - Three persistent scan kernels run the sequential recurrence: U/32
//    workgroups, cell state in registers for all T=512 steps, raw h
//    ping-ponged through global + LDS, one grid barrier per step.
//  - Critical-path tuning: per-step xw loads issued BEFORE the recurrent
//    matmul (latency overlapped), next step's xw prefetched during this step.
//  - ReLU+BN fused into the scan epilogue (bf16 out, feeds next GEMM).
// ---------------------------------------------------------------------------

typedef __attribute__((ext_vector_type(16))) __bf16 bf16x16;
typedef __attribute__((ext_vector_type(8)))  float  f32x8;

#define BB 32
#define TT 512
#define BT (BB * TT)

// ---------------------------------------------------------------- init
__global__ void k_init(unsigned* bar) {
    if (threadIdx.x < 8) bar[threadIdx.x] = 0u;
}

// ---------------------------------------------------------------- f32 -> bf16
__global__ void k_cvt(const float* __restrict__ in, __bf16* __restrict__ out, int n) {
    int i = blockIdx.x * 256 + threadIdx.x;
    if (i < n) out[i] = (__bf16)in[i];
}

// ------------------------------------------------- pack W[K][N] fp32 -> WMMA B frags
// Fragment f = kt*(N/16)+nt.  Element e of lane L holds W[kt*32 + e%8 + 8*(L/16) + 16*(e/8)][nt*16 + L%16].
__global__ void k_packB(const float* __restrict__ w, __bf16* __restrict__ out, int K, int N) {
    int idx = blockIdx.x * 256 + threadIdx.x;
    if (idx >= K * N) return;
    int e    = idx & 15;
    int lane = (idx >> 4) & 31;
    int frag = idx >> 9;
    int ntiles = N >> 4;
    int kt = frag / ntiles, nt = frag - kt * ntiles;
    int k = kt * 32 + (e & 7) + 8 * (lane >> 4) + 16 * (e >> 3);
    int n = nt * 16 + (lane & 15);
    out[idx] = (__bf16)w[(size_t)k * N + n];
}

// ------------------------------------------------- GEMM: C[M][N] = A[M][K]@B + bias
// A bf16 row-major, B pre-packed frags, C fp32.  Block = 256 thr (8 waves),
// tile M128 x N64, K staged through LDS in 32-wide slabs.
__global__ __launch_bounds__(256) void k_gemm(
    const __bf16* __restrict__ A, const __bf16* __restrict__ Bp,
    const float* __restrict__ bias, float* __restrict__ C,
    int M, int N, int K)
{
    __shared__ __align__(16) __bf16 sA[128 * 32];
    const int tid  = threadIdx.x;
    const int w    = tid >> 5;
    const int lane = tid & 31;
    const int lhi  = lane >> 4, llo = lane & 15;
    const int m0 = blockIdx.x * 128;
    const int n0 = blockIdx.y * 64;
    const int ntiles = N >> 4;

    f32x8 acc[4] = {{}, {}, {}, {}};

    for (int kt = 0; kt < (K >> 5); ++kt) {
        { // stage A slab: 128 rows x 32 cols, 16 halves per thread
            int row = tid >> 1;
            int ch  = (tid & 1) * 16;
            const __bf16* src = A + (size_t)(m0 + row) * K + kt * 32 + ch;
            __bf16* dst = &sA[row * 32 + ch];
            ((int4*)dst)[0] = ((const int4*)src)[0];
            ((int4*)dst)[1] = ((const int4*)src)[1];
        }
        __syncthreads();
        bf16x16 af; // A fragment: M = w*16+llo, K = e%8 + 8*lhi + 16*(e/8)
        {
            const __bf16* ap = &sA[(w * 16 + llo) * 32 + 8 * lhi];
            ((int4*)&af)[0] = *(const int4*)ap;
            ((int4*)&af)[1] = *(const int4*)(ap + 16);
        }
#pragma unroll
        for (int nt = 0; nt < 4; ++nt) {
            const bf16x16 bf = *(const bf16x16*)(
                Bp + ((size_t)(kt * ntiles + ((n0 >> 4) + nt)) * 32 + lane) * 16);
            acc[nt] = __builtin_amdgcn_wmma_f32_16x16x32_bf16(
                false, af, false, bf, (short)0, acc[nt], false, false);
        }
        __syncthreads();
    }
#pragma unroll
    for (int nt = 0; nt < 4; ++nt) {
        int col = n0 + nt * 16 + llo;
        float bv = bias ? bias[col] : 0.f;
#pragma unroll
        for (int r = 0; r < 8; ++r) {
            int row = m0 + w * 16 + r + 8 * lhi;
            C[(size_t)row * N + col] = acc[nt][r] + bv;
        }
    }
}

// ------------------------------------------------- persistent LSTM scan
// Grid = U/32 WGs, block = 128 thr (4 waves). Wave: mtile = batch half,
// ttile = 16-col half of this WG's 32 hidden columns; computes all 4 gates.
// c lives in registers across all T steps; h ping-pongs via hbuf + grid barrier.
__global__ __launch_bounds__(128) void k_scan(
    const float* __restrict__ xw, const __bf16* __restrict__ Up,
    __bf16* __restrict__ hbuf, __bf16* __restrict__ seq_out,
    const float* __restrict__ gamma, const float* __restrict__ beta,
    const float* __restrict__ mean,  const float* __restrict__ var,
    float* __restrict__ dout, int U, int T,
    unsigned* bar_cnt, unsigned* bar_gen, int nwg)
{
    __shared__ __align__(16) __bf16 sH[32 * 512];
    const int tid  = threadIdx.x;
    const int w    = tid >> 5;
    const int lane = tid & 31;
    const int lhi  = lane >> 4, llo = lane & 15;
    const int mtile = w & 1;
    const int ttile = (w >> 1) & 1;
    const int g  = blockIdx.x;
    const int N4 = 4 * U;
    const int ntilesN = N4 >> 4;
    const int ktiles  = U >> 5;
    const int u = g * 32 + ttile * 16 + llo;   // hidden column of this lane
    const size_t hsz = (size_t)32 * U;

    float bn_g = 1.f, bn_b = 0.f, bn_m = 0.f, bn_i = 1.f;
    if (gamma) {
        bn_g = gamma[u]; bn_b = beta[u]; bn_m = mean[u];
        bn_i = rsqrtf(var[u] + 1e-3f);
    }

    f32x8 c = {};
    for (int ts = 0; ts < T; ++ts) {
        // ---- issue this step's xw loads FIRST so HBM latency overlaps the
        //      h-staging + WMMA work below; prefetch step ts+1 while we're at it
        float zx0[8], zx1[8], zx2[8], zx3[8];
#pragma unroll
        for (int r = 0; r < 8; ++r) {
            int b = mtile * 16 + r + 8 * lhi;
            size_t base = ((size_t)b * T + ts) * N4;
            zx0[r] = xw[base + u];
            zx1[r] = xw[base + (size_t)U + u];
            zx2[r] = xw[base + 2 * (size_t)U + u];
            zx3[r] = xw[base + 3 * (size_t)U + u];
            if (ts + 1 < T) {   // gfx1250 global_prefetch_b8, no LOADcnt cost
                __builtin_prefetch(&xw[base + N4 + u], 0, 1);
                __builtin_prefetch(&xw[base + N4 + (size_t)U + u], 0, 1);
                __builtin_prefetch(&xw[base + N4 + 2 * (size_t)U + u], 0, 1);
                __builtin_prefetch(&xw[base + N4 + 3 * (size_t)U + u], 0, 1);
            }
        }

        f32x8 zq[4] = {{}, {}, {}, {}};
        if (ts > 0) {
            // stage h_{ts-1} (32 x U bf16) into LDS
            const __bf16* hprev = hbuf + (size_t)((ts - 1) & 1) * hsz;
            const int elems = 32 * U;
            for (int off = tid * 8; off < elems; off += 128 * 8)
                *(int4*)&sH[off] = *(const int4*)&hprev[off];
            __syncthreads();
            for (int kt = 0; kt < ktiles; ++kt) {
                bf16x16 af;
                const __bf16* ap = &sH[(mtile * 16 + llo) * U + kt * 32 + 8 * lhi];
                ((int4*)&af)[0] = *(const int4*)ap;
                ((int4*)&af)[1] = *(const int4*)(ap + 16);
#pragma unroll
                for (int q = 0; q < 4; ++q) {
                    int ntile = q * (U >> 4) + g * 2 + ttile;
                    const bf16x16 bf = *(const bf16x16*)(
                        Up + ((size_t)(kt * ntilesN + ntile) * 32 + lane) * 16);
                    zq[q] = __builtin_amdgcn_wmma_f32_16x16x32_bf16(
                        false, af, false, bf, (short)0, zq[q], false, false);
                }
            }
            __syncthreads();  // done reading sH before next step's restage
        }
        // gates + state update + outputs
        __bf16* hcur = hbuf + (size_t)(ts & 1) * hsz;
#pragma unroll
        for (int r = 0; r < 8; ++r) {
            int b = mtile * 16 + r + 8 * lhi;
            float zi = zq[0][r] + zx0[r];
            float zf = zq[1][r] + zx1[r];
            float zg = zq[2][r] + zx2[r];
            float zo = zq[3][r] + zx3[r];
            float iv = 1.f / (1.f + __expf(-zi));
            float fv = 1.f / (1.f + __expf(-zf));
            float ov = 1.f / (1.f + __expf(-zo));
            float cv = fv * c[r] + iv * zg;   // candidate activation = identity
            c[r] = cv;
            float hv = ov * cv;               // output activation = identity
            hcur[(size_t)b * U + u] = (__bf16)hv;
            if (seq_out) {
                float pv = hv > 0.f ? hv : 0.f;          // external ReLU
                pv = (pv - bn_m) * bn_i * bn_g + bn_b;   // BN inference
                seq_out[((size_t)b * T + ts) * U + u] = (__bf16)pv;
            }
            if (dout && ts == T - 1) dout[b * U + u] = hv;
        }
        // ---- grid-wide sense-reversing barrier (all WGs resident) ----
        __threadfence();
        __syncthreads();
        if (tid == 0) {
            unsigned gen = __hip_atomic_load(bar_gen, __ATOMIC_ACQUIRE,
                                             __HIP_MEMORY_SCOPE_AGENT);
            unsigned arrived = __hip_atomic_fetch_add(bar_cnt, 1u, __ATOMIC_ACQ_REL,
                                                      __HIP_MEMORY_SCOPE_AGENT);
            if (arrived == (unsigned)(nwg - 1)) {
                __hip_atomic_store(bar_cnt, 0u, __ATOMIC_RELAXED,
                                   __HIP_MEMORY_SCOPE_AGENT);
                __hip_atomic_fetch_add(bar_gen, 1u, __ATOMIC_RELEASE,
                                       __HIP_MEMORY_SCOPE_AGENT);
            } else {
                while (__hip_atomic_load(bar_gen, __ATOMIC_ACQUIRE,
                                         __HIP_MEMORY_SCOPE_AGENT) == gen)
                    __builtin_amdgcn_s_sleep(1);
            }
        }
        __syncthreads();
    }
}

// ---------------------------------------------------------------------------
extern "C" void kernel_launch(void* const* d_in, const int* in_sizes, int n_in,
                              void* d_out, int out_size, void* d_ws, size_t ws_size,
                              hipStream_t stream) {
    const float* x  = (const float*)d_in[0];
    const float* W0 = (const float*)d_in[1];  const float* U0 = (const float*)d_in[2];
    const float* b0 = (const float*)d_in[3];
    const float* g0 = (const float*)d_in[4];  const float* be0 = (const float*)d_in[5];
    const float* m0 = (const float*)d_in[6];  const float* v0  = (const float*)d_in[7];
    const float* W1 = (const float*)d_in[8];  const float* U1 = (const float*)d_in[9];
    const float* b1 = (const float*)d_in[10];
    const float* g1 = (const float*)d_in[11]; const float* be1 = (const float*)d_in[12];
    const float* m1 = (const float*)d_in[13]; const float* v1  = (const float*)d_in[14];
    const float* Wl = (const float*)d_in[15]; const float* Ul = (const float*)d_in[16];
    const float* bl = (const float*)d_in[17];
    float* out = (float*)d_out;

    // -------- workspace carve-out (all 256B aligned), ~175 MB total --------
    char* ws = (char*)d_ws;
    size_t off = 0;
    auto carve = [&](size_t bytes) -> char* {
        off = (off + 255) & ~(size_t)255;
        char* p = ws + off;
        off += bytes;
        return p;
    };
    unsigned* bar  = (unsigned*)carve(256);            // [0]=count, [1]=gen
    __bf16* xbf  = (__bf16*)carve((size_t)BT * 256 * 2);
    __bf16* seq0 = (__bf16*)carve((size_t)BT * 512 * 2);
    __bf16* seq1 = (__bf16*)carve((size_t)BT * 512 * 2);
    float*  xw   = (float*) carve((size_t)BT * 2048 * 4);   // reused by all layers
    __bf16* W0p  = (__bf16*)carve((size_t)256 * 2048 * 2);
    __bf16* U0p  = (__bf16*)carve((size_t)512 * 2048 * 2);
    __bf16* W1p  = (__bf16*)carve((size_t)512 * 2048 * 2);
    __bf16* U1p  = (__bf16*)carve((size_t)512 * 2048 * 2);
    __bf16* Wlp  = (__bf16*)carve((size_t)512 * 1024 * 2);
    __bf16* Ulp  = (__bf16*)carve((size_t)256 * 1024 * 2);
    __bf16* hbuf = (__bf16*)carve((size_t)2 * 32 * 512 * 2);
    (void)ws_size; (void)in_sizes; (void)n_in; (void)out_size;

    auto blocks = [](int n) { return (n + 255) / 256; };

    k_init<<<1, 64, 0, stream>>>(bar);
    // input + weight precision conversion / packing (weights stay hot in L2)
    k_cvt<<<blocks(BT * 256), 256, 0, stream>>>(x, xbf, BT * 256);
    k_packB<<<blocks(256 * 2048), 256, 0, stream>>>(W0, W0p, 256, 2048);
    k_packB<<<blocks(512 * 2048), 256, 0, stream>>>(U0, U0p, 512, 2048);
    k_packB<<<blocks(512 * 2048), 256, 0, stream>>>(W1, W1p, 512, 2048);
    k_packB<<<blocks(512 * 2048), 256, 0, stream>>>(U1, U1p, 512, 2048);
    k_packB<<<blocks(512 * 1024), 256, 0, stream>>>(Wl, Wlp, 512, 1024);
    k_packB<<<blocks(256 * 1024), 256, 0, stream>>>(Ul, Ulp, 256, 1024);

    // ---- layer 0 ----
    k_gemm<<<dim3(BT / 128, 2048 / 64), 256, 0, stream>>>(xbf, W0p, b0, xw, BT, 2048, 256);
    k_scan<<<16, 128, 0, stream>>>(xw, U0p, hbuf, seq0, g0, be0, m0, v0,
                                   nullptr, 512, TT, bar, bar + 1, 16);
    // ---- layer 1 ----
    k_gemm<<<dim3(BT / 128, 2048 / 64), 256, 0, stream>>>(seq0, W1p, b1, xw, BT, 2048, 512);
    k_scan<<<16, 128, 0, stream>>>(xw, U1p, hbuf, seq1, g1, be1, m1, v1,
                                   nullptr, 512, TT, bar, bar + 1, 16);
    // ---- latent layer (return last hidden state, fp32) ----
    k_gemm<<<dim3(BT / 128, 1024 / 64), 256, 0, stream>>>(seq1, Wlp, bl, xw, BT, 1024, 512);
    k_scan<<<8, 128, 0, stream>>>(xw, Ulp, hbuf, nullptr, nullptr, nullptr, nullptr,
                                  nullptr, out, 256, TT, bar, bar + 1, 8);
}
